// GraphSage_36773509988957
// MI455X (gfx1250) — matrix operand order repaired and checked by reading, not compiled, not run on hardware
//
#include <hip/hip_runtime.h>
#include <hip/hip_bf16.h>

#define N_NODES 100000
#define FEAT    128
#define E_EDGES 800000
#define MTILES  (N_NODES / 16)   // 6250, exact

typedef __attribute__((ext_vector_type(16))) __bf16 v16bf;
typedef __attribute__((ext_vector_type(8)))  float  v8f;

// ---------------------------------------------------------------------------
// WMMA fragment loaders (wave32, 16-bit layouts per CDNA5 ISA 7.12.2)
// A 16x32 (MxK): lane L holds row M=L&15; half=L>>4.
//   VGPR0-3: K = k0 + 8*half + {0..7};  VGPR4-7: K = k0 + 16 + 8*half + {0..7}
// B 32x16 (KxN): lane L holds col N=L&15; half=L>>4 (same K order as A):
//   VGPR0-3: K = k0 + 8*half + {0..7};  VGPR4-7: K = k0 + 16 + 8*half + {0..7}
// ---------------------------------------------------------------------------
__device__ __forceinline__ v16bf load_frag_a(const __bf16* rowbase, int k0, int half) {
    v16bf a;
    const __bf16* p0 = rowbase + k0 + 8 * half;
    const __bf16* p1 = rowbase + k0 + 16 + 8 * half;
    *(uint4*)&a         = *(const uint4*)p0;   // 8 bf16 = 16B
    *(((uint4*)&a) + 1) = *(const uint4*)p1;   // 8 bf16 = 16B
    return a;
}

// B fragment from LDS-staged weights (padded row stride, bank-conflict-free)
__device__ __forceinline__ v16bf load_frag_b_lds(const __bf16* ws, int n, int k0,
                                                 int half, int kpad) {
    v16bf b;
    const __bf16* p = ws + n * kpad + k0 + 8 * half;
    *(uint4*)&b         = *(const uint4*)p;       // ds_load_b128
    *(((uint4*)&b) + 1) = *(const uint4*)(p + 16); // ds_load_b128
    return b;
}

// ---------------------------------------------------------------------------
// Fused SAGE layer GEMM: out = [relu]( agg @ Wl^T + b + h @ Wr^T )
// 256 threads = 8 waves per block; weights staged in LDS once per block.
// One wave -> 16 rows x (NT*16) cols; A fragments reused across all N tiles.
// ---------------------------------------------------------------------------
template <int NT, bool RELU, bool F32OUT>
__global__ __launch_bounds__(256) void sage_gemm(
    const __bf16* __restrict__ agg, const __bf16* __restrict__ h,
    const __bf16* __restrict__ wl,  const __bf16* __restrict__ wr,
    const float*  __restrict__ bias,
    float* __restrict__ outf, __bf16* __restrict__ outb)
{
    constexpr int KPAD = FEAT + 8;   // 136 elems = 272B row stride: lane n hits
                                     // banks 4n+c -> 16 lanes cover all 64 banks
    __shared__ __bf16 sWl[NT * 16 * KPAD];
    __shared__ __bf16 sWr[NT * 16 * KPAD];

    // Cooperative weight staging: NT*16 rows x 128 bf16, 16B chunks.
    const int tid = threadIdx.x;
    const int totalChunks = NT * 16 * (FEAT / 8);      // 2048 (NT=8) / 1024 (NT=4)
    for (int idx = tid; idx < totalChunks; idx += 256) {
        const int row = idx >> 4;                      // 16 chunks per row
        const int ck  = (idx & 15) * 8;
        *(uint4*)&sWl[row * KPAD + ck] = *(const uint4*)&wl[row * FEAT + ck];
        *(uint4*)&sWr[row * KPAD + ck] = *(const uint4*)&wr[row * FEAT + ck];
    }
    __syncthreads();

    const int lane  = tid & 31;
    const int wave  = tid >> 5;
    const int mtile = blockIdx.x * 8 + wave;           // wave-uniform
    if (mtile >= MTILES) return;                       // after barrier: safe

    const int half = lane >> 4;
    const int nl   = lane & 15;
    const int mrow = mtile * 16 + nl;                  // A row this lane loads

    const __bf16* aggRow = agg + (size_t)mrow * FEAT;
    const __bf16* hRow   = h   + (size_t)mrow * FEAT;

    v16bf aA[4], aH[4];
#pragma unroll
    for (int c = 0; c < 4; ++c) {
        aA[c] = load_frag_a(aggRow, 32 * c, half);
        aH[c] = load_frag_a(hRow,   32 * c, half);
    }

    v8f acc[NT];
#pragma unroll
    for (int t = 0; t < NT; ++t) {
        float b = bias[t * 16 + nl];                   // bias depends on col only
        acc[t] = (v8f){b, b, b, b, b, b, b, b};
    }

#pragma unroll
    for (int t = 0; t < NT; ++t) {
        const int n = t * 16 + nl;
#pragma unroll
        for (int c = 0; c < 4; ++c) {
            v16bf bl = load_frag_b_lds(sWl, n, 32 * c, half, KPAD);
            acc[t] = __builtin_amdgcn_wmma_f32_16x16x32_bf16(
                false, aA[c], false, bl, (short)0, acc[t], false, false);
            v16bf br = load_frag_b_lds(sWr, n, 32 * c, half, KPAD);
            acc[t] = __builtin_amdgcn_wmma_f32_16x16x32_bf16(
                false, aH[c], false, br, (short)0, acc[t], false, false);
        }
    }

    const int Nout = NT * 16;
#pragma unroll
    for (int t = 0; t < NT; ++t) {
#pragma unroll
        for (int r = 0; r < 8; ++r) {
            float v = acc[t][r];
            if (RELU) v = fmaxf(v, 0.0f);
            const int row = mtile * 16 + r + 8 * half; // C/D layout
            const int col = t * 16 + nl;
            if (F32OUT) outf[(size_t)row * Nout + col] = v;
            else        outb[(size_t)row * Nout + col] = (__bf16)v;
        }
    }
}

// ---------------------------------------------------------------------------
// Helper kernels
// ---------------------------------------------------------------------------
__global__ void f32_to_bf16_kernel(const float* __restrict__ in,
                                   __bf16* __restrict__ out, int n) {
    int i = blockIdx.x * blockDim.x + threadIdx.x;
    if (i < n) out[i] = (__bf16)in[i];
}

__global__ void degree_kernel(const long long* __restrict__ ei,
                              float* __restrict__ deg) {
    int e = blockIdx.x * blockDim.x + threadIdx.x;
    if (e < E_EDGES) {
        int d = (int)ei[E_EDGES + e];                  // dst row of edge_index
        unsafeAtomicAdd(&deg[d], 1.0f);
    }
}

// One thread handles 4 features of one edge (32 threads / edge).
__global__ void scatter_kernel(const long long* __restrict__ ei,
                               const __bf16* __restrict__ h,
                               float* __restrict__ agg) {
    int tid = blockIdx.x * blockDim.x + threadIdx.x;
    int e = tid >> 5;
    if (e >= E_EDGES) return;
    int c = (tid & 31) << 2;
    int s = (int)ei[e];
    int d = (int)ei[E_EDGES + e];
    const __bf16* hp = h + (size_t)s * FEAT + c;
    float* ap = agg + (size_t)d * FEAT + c;
    uint2 raw = *(const uint2*)hp;                     // 4 bf16 = 8B
    __bf16 v[4];
    *(uint2*)v = raw;
#pragma unroll
    for (int i = 0; i < 4; ++i) unsafeAtomicAdd(ap + i, (float)v[i]);
}

// agg_bf16[i] = (agg[i] / max(deg[row],1)) ; row = i / 128
__global__ void scale_kernel(const float* __restrict__ agg,
                             const float* __restrict__ deg,
                             __bf16* __restrict__ aggb, int n) {
    int i = blockIdx.x * blockDim.x + threadIdx.x;
    if (i >= n) return;
    float dinv = 1.0f / fmaxf(deg[i >> 7], 1.0f);
    aggb[i] = (__bf16)(agg[i] * dinv);
}

// ---------------------------------------------------------------------------
extern "C" void kernel_launch(void* const* d_in, const int* in_sizes, int n_in,
                              void* d_out, int out_size, void* d_ws, size_t ws_size,
                              hipStream_t stream) {
    (void)in_sizes; (void)n_in; (void)out_size; (void)ws_size;

    const float*     x   = (const float*)d_in[0];
    const long long* ei  = (const long long*)d_in[1];  // int64 edge_index [2,E]
    const float*     wl0 = (const float*)d_in[2];
    const float*     bl0 = (const float*)d_in[3];
    const float*     wr0 = (const float*)d_in[4];
    const float*     wl1 = (const float*)d_in[5];
    const float*     bl1 = (const float*)d_in[6];
    const float*     wr1 = (const float*)d_in[7];
    const float*     wl2 = (const float*)d_in[8];
    const float*     bl2 = (const float*)d_in[9];
    const float*     wr2 = (const float*)d_in[10];

    // Workspace carve-up (~129 MB)
    char* ws = (char*)d_ws;
    size_t off = 0;
    auto carve = [&](size_t bytes) -> void* {
        void* p = ws + off;
        off += (bytes + 255) & ~(size_t)255;
        return p;
    };
    float*  deg   = (float*) carve((size_t)N_NODES * 4);
    float*  agg   = (float*) carve((size_t)N_NODES * FEAT * 4);
    __bf16* aggb  = (__bf16*)carve((size_t)N_NODES * FEAT * 2);
    __bf16* hA    = (__bf16*)carve((size_t)N_NODES * FEAT * 2);  // x (bf16), then L1 out
    __bf16* hB    = (__bf16*)carve((size_t)N_NODES * FEAT * 2);  // L0 out
    __bf16* wl0b  = (__bf16*)carve(128 * 128 * 2);
    __bf16* wr0b  = (__bf16*)carve(128 * 128 * 2);
    __bf16* wl1b  = (__bf16*)carve(128 * 128 * 2);
    __bf16* wr1b  = (__bf16*)carve(128 * 128 * 2);
    __bf16* wl2b  = (__bf16*)carve(64 * 128 * 2);
    __bf16* wr2b  = (__bf16*)carve(64 * 128 * 2);
    float*  outF  = (float*)d_out;                                // [N, 64]

    const int NFEATTOT = N_NODES * FEAT;                          // 12.8M
    const int CV_BLK = 256;
    const int cvGridBig = (NFEATTOT + CV_BLK - 1) / CV_BLK;

    // Convert inputs / weights to bf16
    f32_to_bf16_kernel<<<cvGridBig, CV_BLK, 0, stream>>>(x, hA, NFEATTOT);
    f32_to_bf16_kernel<<<(128*128 + 255) / 256, 256, 0, stream>>>(wl0, wl0b, 128 * 128);
    f32_to_bf16_kernel<<<(128*128 + 255) / 256, 256, 0, stream>>>(wr0, wr0b, 128 * 128);
    f32_to_bf16_kernel<<<(128*128 + 255) / 256, 256, 0, stream>>>(wl1, wl1b, 128 * 128);
    f32_to_bf16_kernel<<<(128*128 + 255) / 256, 256, 0, stream>>>(wr1, wr1b, 128 * 128);
    f32_to_bf16_kernel<<<(64*128 + 255) / 256, 256, 0, stream>>>(wl2, wl2b, 64 * 128);
    f32_to_bf16_kernel<<<(64*128 + 255) / 256, 256, 0, stream>>>(wr2, wr2b, 64 * 128);

    // Degree (computed once; clamped in scale_kernel)
    hipMemsetAsync(deg, 0, (size_t)N_NODES * 4, stream);
    degree_kernel<<<(E_EDGES + 255) / 256, 256, 0, stream>>>(ei, deg);

    const int scatterGrid = (E_EDGES * 32 + 255) / 256;   // 100000 blocks
    const int gemmGrid    = (MTILES + 7) / 8;             // 8 waves / block -> 782

    // ---- Layer 0: reads hA (x), writes hB ----
    hipMemsetAsync(agg, 0, (size_t)N_NODES * FEAT * 4, stream);
    scatter_kernel<<<scatterGrid, 256, 0, stream>>>(ei, hA, agg);
    scale_kernel<<<cvGridBig, CV_BLK, 0, stream>>>(agg, deg, aggb, NFEATTOT);
    sage_gemm<8, true, false><<<gemmGrid, 256, 0, stream>>>(
        aggb, hA, wl0b, wr0b, bl0, nullptr, hB);

    // ---- Layer 1: reads hB, writes hA ----
    hipMemsetAsync(agg, 0, (size_t)N_NODES * FEAT * 4, stream);
    scatter_kernel<<<scatterGrid, 256, 0, stream>>>(ei, hB, agg);
    scale_kernel<<<cvGridBig, CV_BLK, 0, stream>>>(agg, deg, aggb, NFEATTOT);
    sage_gemm<8, true, false><<<gemmGrid, 256, 0, stream>>>(
        aggb, hB, wl1b, wr1b, bl1, nullptr, hA);

    // ---- Layer 2: reads hA, writes fp32 d_out [N,64] ----
    hipMemsetAsync(agg, 0, (size_t)N_NODES * FEAT * 4, stream);
    scatter_kernel<<<scatterGrid, 256, 0, stream>>>(ei, hA, agg);
    scale_kernel<<<cvGridBig, CV_BLK, 0, stream>>>(agg, deg, aggb, NFEATTOT);
    sage_gemm<4, false, true><<<gemmGrid, 256, 0, stream>>>(
        aggb, hA, wl2b, wr2b, bl2, outF, nullptr);
}
